// PointCNN_48077863911901
// MI455X (gfx1250) — compile-verified, matrix operand order checked by reading, not run on hardware
//
#include <hip/hip_runtime.h>
#include <hip/hip_bf16.h>
#include <math.h>

typedef __attribute__((ext_vector_type(16))) _Float16 v16h;
typedef __attribute__((ext_vector_type(8)))  float    v8f;

// ---------------------------------------------------------------------------
// kNN: one thread per (b, m). Insertion-sorted top-(k*dil) list, then pick
// every dil-th entry (matches top_k(-d2)[:, :, ::dilation]).
// All layer coords are prefixes of the original p (B, 2048, 3).
// ---------------------------------------------------------------------------
#define MAXKD 48
__global__ void knn_kernel(const float* __restrict__ p, int N, int Mq, int B,
                           int k, int dil, int* __restrict__ idxOut) {
    int t = blockIdx.x * blockDim.x + threadIdx.x;
    if (t >= B * Mq) return;
    int b = t / Mq, m = t % Mq;
    const float* Pb = p + (size_t)b * 2048 * 3;
    float qx = Pb[m * 3 + 0], qy = Pb[m * 3 + 1], qz = Pb[m * 3 + 2];
    int KD = k * dil;
    float bd[MAXKD];
    int   bi[MAXKD];
    for (int i = 0; i < KD; ++i) { bd[i] = 3.4e38f; bi[i] = 0; }
    for (int n = 0; n < N; ++n) {
        float dx = Pb[n * 3 + 0] - qx;
        float dy = Pb[n * 3 + 1] - qy;
        float dz = Pb[n * 3 + 2] - qz;
        float d = dx * dx + dy * dy + dz * dz;
        if (d < bd[KD - 1]) {
            int j = KD - 1;
            while (j > 0 && bd[j - 1] > d) {
                bd[j] = bd[j - 1]; bi[j] = bi[j - 1]; --j;
            }
            bd[j] = d; bi[j] = n;
        }
    }
    for (int j = 0; j < k; ++j) idxOut[(size_t)t * k + j] = bi[j * dil];
}

// ---------------------------------------------------------------------------
// Gather: pl = p[idx] - q  (B*M*k, 3); gathered x features go into the
// concat buffer hcat[:, mid:mid+cin].
// ---------------------------------------------------------------------------
__global__ void gather_kernel(const float* __restrict__ p, const int* __restrict__ idx,
                              const float* __restrict__ x, int Nx, int cin,
                              int Mq, int B, int k, int Cc, int mid,
                              float* __restrict__ pl, float* __restrict__ hcat) {
    int t = blockIdx.x * blockDim.x + threadIdx.x;
    int rows = B * Mq * k;
    if (t >= rows) return;
    int bm = t / k;
    int b = bm / Mq, m = bm % Mq;
    int n = idx[t];
    const float* Pb = p + (size_t)b * 2048 * 3;
    pl[(size_t)t * 3 + 0] = Pb[n * 3 + 0] - Pb[m * 3 + 0];
    pl[(size_t)t * 3 + 1] = Pb[n * 3 + 1] - Pb[m * 3 + 1];
    pl[(size_t)t * 3 + 2] = Pb[n * 3 + 2] - Pb[m * 3 + 2];
    if (x != nullptr) {
        const float* xr = x + ((size_t)b * Nx + n) * cin;
        float* hr = hcat + (size_t)t * Cc + mid;
        for (int c = 0; c < cin; ++c) hr[c] = xr[c];
    }
}

// ---------------------------------------------------------------------------
// K=3 dense layer with fused BN + ELU: y = ELU(BN(pl @ W(3xN) + b)).
// ---------------------------------------------------------------------------
__global__ void dense3_kernel(const float* __restrict__ in, const float* __restrict__ W,
                              const float* __restrict__ bias,
                              const float* __restrict__ g, const float* __restrict__ bb,
                              const float* __restrict__ mm, const float* __restrict__ vv,
                              float* __restrict__ out, long total, int N) {
    long t = (long)blockIdx.x * blockDim.x + threadIdx.x;
    long stride = (long)gridDim.x * blockDim.x;
    for (; t < total; t += stride) {
        long r = t / N;
        int  c = (int)(t % N);
        float v0 = in[r * 3 + 0], v1 = in[r * 3 + 1], v2 = in[r * 3 + 2];
        float y = v0 * W[c] + v1 * W[N + c] + v2 * W[2 * N + c] + bias[c];
        float inv = rsqrtf(vv[c] + 1e-5f);
        y = (y - mm[c]) * inv * g[c] + bb[c];
        y = (y > 0.0f) ? y : (expf(y) - 1.0f);   // ELU
        out[t] = y;
    }
}

// ---------------------------------------------------------------------------
// Async global -> LDS copy of 16 bytes (CDNA5 async path, ASYNCcnt).
// ---------------------------------------------------------------------------
__device__ __forceinline__ void async_copy_b128(void* lds_ptr, const void* gsrc) {
    unsigned lo = (unsigned)(uintptr_t)lds_ptr;
    asm volatile("global_load_async_to_lds_b128 %0, %1, off"
                 :: "v"(lo), "v"(gsrc) : "memory");
}
__device__ __forceinline__ void wait_asynccnt0() {
    asm volatile("s_wait_asynccnt 0" ::: "memory");
}

// ---------------------------------------------------------------------------
// WMMA GEMM tile config: 64x64 block tile, 4 waves, 32-deep K chunks.
// LDS rows padded to 36 floats (144B): conflict-free ds_load_b128 fragment
// fetches, still 16B-aligned for async b128 stores.
// ---------------------------------------------------------------------------
#define BMT 64
#define BNT 64
#define BKT 32
#define LSTR 36

// Fragment layout per ISA 7.12.2 (16-bit A 16x32): halves 0..7 = K khalf..+7,
// halves 8..15 = K 16+khalf..+23. Built from f32 LDS rows with float4 loads.
__device__ __forceinline__ v16h load_frag(const float* s, int khalf) {
    float4 a = *(const float4*)(s + khalf);
    float4 b = *(const float4*)(s + khalf + 4);
    float4 c = *(const float4*)(s + khalf + 16);
    float4 d = *(const float4*)(s + khalf + 20);
    v16h f;
    f[0]  = (_Float16)a.x; f[1]  = (_Float16)a.y; f[2]  = (_Float16)a.z; f[3]  = (_Float16)a.w;
    f[4]  = (_Float16)b.x; f[5]  = (_Float16)b.y; f[6]  = (_Float16)b.z; f[7]  = (_Float16)b.w;
    f[8]  = (_Float16)c.x; f[9]  = (_Float16)c.y; f[10] = (_Float16)c.z; f[11] = (_Float16)c.w;
    f[12] = (_Float16)d.x; f[13] = (_Float16)d.y; f[14] = (_Float16)d.z; f[15] = (_Float16)d.w;
    return f;
}

__device__ __forceinline__ v8f wmma_f16(v16h a, v16h b, v8f c) {
    return __builtin_amdgcn_wmma_f32_16x16x32_f16(false, a, false, b, (short)0, c,
                                                  false, false);
}

// Stage a 64x32 f32 tile (rows of src, layout [r][k] in LDS, stride LSTR).
// Interior 16B chunks go through the async engine; edges take guarded loads.
__device__ __forceinline__ void stage_rowmajor(float* sbuf, const float* __restrict__ src,
                                               int ld, int row0, int k0,
                                               int RMAX, int K, int tid) {
#pragma unroll
    for (int e = 0; e < 4; ++e) {
        int ci = tid + e * 128;
        int r  = ci >> 3;              // 8 chunks of 4 floats per 32-float row
        int c4 = (ci & 7) * 4;
        float* dst = &sbuf[r * LSTR + c4];
        int gr = row0 + r;
        if (gr < RMAX && k0 + c4 + 4 <= K) {
            async_copy_b128(dst, src + (size_t)gr * ld + (k0 + c4));
        } else {
            float4 v = {0.0f, 0.0f, 0.0f, 0.0f};
            if (gr < RMAX) {
                if (k0 + c4 + 0 < K) v.x = src[(size_t)gr * ld + k0 + c4 + 0];
                if (k0 + c4 + 1 < K) v.y = src[(size_t)gr * ld + k0 + c4 + 1];
                if (k0 + c4 + 2 < K) v.z = src[(size_t)gr * ld + k0 + c4 + 2];
                if (k0 + c4 + 3 < K) v.w = src[(size_t)gr * ld + k0 + c4 + 3];
            }
            *(float4*)dst = v;
        }
    }
}

// Non-transposed B (K x N): prefetch float4 chunks along N into registers
// (coalesced), scattered later into the transposed-in-LDS [n][k] layout.
struct BReg { float4 v[4]; };

__device__ __forceinline__ BReg loadB_nt(const float* __restrict__ Bm, int ldb,
                                         int blockCol, int k0, int N, int K, int tid) {
    BReg rg;
#pragma unroll
    for (int e = 0; e < 4; ++e) {
        int ci = tid + e * 128;
        int kx = ci >> 4;              // 16 chunks of 4 cols per k-row
        int n4 = (ci & 15) * 4;
        float4 v = {0.0f, 0.0f, 0.0f, 0.0f};
        int gk = k0 + kx;
        int gn = blockCol + n4;
        if (gk < K) {
            if (gn + 4 <= N) {
                v = *(const float4*)&Bm[(size_t)gk * ldb + gn];
            } else {
                if (gn + 0 < N) v.x = Bm[(size_t)gk * ldb + gn + 0];
                if (gn + 1 < N) v.y = Bm[(size_t)gk * ldb + gn + 1];
                if (gn + 2 < N) v.z = Bm[(size_t)gk * ldb + gn + 2];
                if (gn + 3 < N) v.w = Bm[(size_t)gk * ldb + gn + 3];
            }
        }
        rg.v[e] = v;
    }
    return rg;
}

__device__ __forceinline__ void storeB_nt(float* sbuf, const BReg& rg, int tid) {
#pragma unroll
    for (int e = 0; e < 4; ++e) {
        int ci = tid + e * 128;
        int kx = ci >> 4;
        int n4 = (ci & 15) * 4;
        sbuf[(n4 + 0) * LSTR + kx] = rg.v[e].x;
        sbuf[(n4 + 1) * LSTR + kx] = rg.v[e].y;
        sbuf[(n4 + 2) * LSTR + kx] = rg.v[e].z;
        sbuf[(n4 + 3) * LSTR + kx] = rg.v[e].w;
    }
}

// ---------------------------------------------------------------------------
// Generic WMMA GEMM: C = act(BN(A(MxK) @ B(KxN) + bias)).
// Software-pipelined, double-buffered LDS: chunk i+1's async copies (and
// non-trans B register prefetch) are issued before chunk i's WMMAs;
// s_wait_asynccnt 0 + barrier close each iteration.
// btrans=1: B stored (N, K) row-major (conv_w).  act: 0=none,1=relu,2=elu.
// ---------------------------------------------------------------------------
__global__ void __launch_bounds__(128)
wmma_gemm_kernel(const float* __restrict__ A, int lda,
                 const float* __restrict__ Bm, int ldb, int btrans,
                 const float* __restrict__ bias,
                 const float* __restrict__ bng, const float* __restrict__ bnb,
                 const float* __restrict__ bnm, const float* __restrict__ bnv,
                 int act,
                 float* __restrict__ C, int ldc,
                 int M, int N, int K) {
    __shared__ float sA[2][BMT * LSTR];
    __shared__ float sB[2][BNT * LSTR];
    const int tid  = threadIdx.x;
    const int lane = tid & 31;
    const int wave = tid >> 5;
    const int rw   = wave * 16;            // wave's row offset in block tile
    const int blockRow = blockIdx.y * BMT;
    const int blockCol = blockIdx.x * BNT;
    const bool asyncB  = (btrans != 0);

    v8f acc0 = {}, acc1 = {}, acc2 = {}, acc3 = {};
    BReg breg;

    const int nch = (K + BKT - 1) / BKT;

    // ---- prologue: stage chunk 0 into buffer 0 ----
    stage_rowmajor(sA[0], A, lda, blockRow, 0, M, K, tid);
    if (asyncB) {
        stage_rowmajor(sB[0], Bm, ldb, blockCol, 0, N, K, tid);
    } else {
        breg = loadB_nt(Bm, ldb, blockCol, 0, N, K, tid);
        storeB_nt(sB[0], breg, tid);
    }
    wait_asynccnt0();
    __syncthreads();

    for (int ch = 0; ch < nch; ++ch) {
        const int cur = ch & 1;
        const int nxt = cur ^ 1;
        const bool havenext = (ch + 1 < nch);
        const int k0n = (ch + 1) * BKT;

        // ---- issue next chunk's loads (overlaps with WMMAs below) ----
        if (havenext) {
            stage_rowmajor(sA[nxt], A, lda, blockRow, k0n, M, K, tid);
            if (asyncB) stage_rowmajor(sB[nxt], Bm, ldb, blockCol, k0n, N, K, tid);
            else        breg = loadB_nt(Bm, ldb, blockCol, k0n, N, K, tid);
        }

        // ---- fragments + 4 WMMAs per wave on current buffer ----
        const int mr    = lane & 15;
        const int khalf = (lane & 16) ? 8 : 0;
        const float* a_ = sA[cur];
        const float* b_ = sB[cur];
        v16h af = load_frag(&a_[(rw + mr) * LSTR], khalf);
        v16h b0 = load_frag(&b_[(0  + mr) * LSTR], khalf);
        v16h b1 = load_frag(&b_[(16 + mr) * LSTR], khalf);
        v16h b2 = load_frag(&b_[(32 + mr) * LSTR], khalf);
        v16h b3 = load_frag(&b_[(48 + mr) * LSTR], khalf);
        acc0 = wmma_f16(af, b0, acc0);
        acc1 = wmma_f16(af, b1, acc1);
        acc2 = wmma_f16(af, b2, acc2);
        acc3 = wmma_f16(af, b3, acc3);

        if (havenext && !asyncB) storeB_nt(sB[nxt], breg, tid);
        wait_asynccnt0();
        __syncthreads();
    }

    // ---- epilogue: C/D layout = VGPR r -> M = r + (lane>=16 ? 8 : 0) ----
    const int n0      = lane & 15;
    const int rowBase = blockRow + rw + ((lane & 16) ? 8 : 0);
#pragma unroll
    for (int ct = 0; ct < 4; ++ct) {
        v8f acc = (ct == 0) ? acc0 : (ct == 1) ? acc1 : (ct == 2) ? acc2 : acc3;
        int col = blockCol + ct * 16 + n0;
        if (col < N) {
            float bsv = bias ? bias[col] : 0.0f;
            float sc = 1.0f, sh = 0.0f;
            if (bng) {
                float inv = rsqrtf(bnv[col] + 1e-5f);
                sc = bng[col] * inv;
                sh = bnb[col] - bnm[col] * sc;
            }
#pragma unroll
            for (int r = 0; r < 8; ++r) {
                int row = rowBase + r;
                if (row < M) {
                    float y = acc[r] + bsv;
                    y = y * sc + sh;
                    if (act == 1)      y = fmaxf(y, 0.0f);
                    else if (act == 2) y = (y > 0.0f) ? y : (expf(y) - 1.0f);
                    C[(size_t)row * ldc + col] = y;
                }
            }
        }
    }
}

// ---------------------------------------------------------------------------
// Apply per-point k x k transform: htr[bm,i,c] = sum_j T[bm,i,j]*hcat[bm,j,c]
// ---------------------------------------------------------------------------
__global__ void trans_kernel(const float* __restrict__ T, const float* __restrict__ hcat,
                             float* __restrict__ htr, int bmTot, int k, int Cc) {
    long t = (long)blockIdx.x * blockDim.x + threadIdx.x;
    long total = (long)bmTot * k * Cc;
    if (t >= total) return;
    int c  = (int)(t % Cc);
    int i  = (int)((t / Cc) % k);
    long bm = t / ((long)Cc * k);
    const float* Tr = T + bm * (k * k) + (size_t)i * k;
    const float* H  = hcat + bm * (size_t)k * Cc;
    float s = 0.0f;
    for (int j = 0; j < k; ++j) s += Tr[j] * H[(size_t)j * Cc + c];
    htr[t] = s;
}

// ---------------------------------------------------------------------------
// Mean over points: out[b, o] = mean_m logits[b*Mp + m, o]
// ---------------------------------------------------------------------------
__global__ void mean_kernel(const float* __restrict__ logits, float* __restrict__ out,
                            int B, int Mp, int C) {
    int t = blockIdx.x * blockDim.x + threadIdx.x;
    if (t >= B * C) return;
    int b = t / C, c = t % C;
    float s = 0.0f;
    for (int m = 0; m < Mp; ++m) s += logits[((size_t)b * Mp + m) * C + c];
    out[t] = s / (float)Mp;
}

// ---------------------------------------------------------------------------
// Host orchestration
// ---------------------------------------------------------------------------
extern "C" void kernel_launch(void* const* d_in, const int* in_sizes, int n_in,
                              void* d_out, int out_size, void* d_ws, size_t ws_size,
                              hipStream_t stream) {
    (void)in_sizes; (void)n_in; (void)out_size; (void)ws_size;
    const float* P = (const float*)d_in[0];  // (32, 2048, 3)
    const int B = 32;
    auto FP = [&](int i) { return (const float*)d_in[i]; };

    // Workspace bump allocator (sizes = max over layers)
    size_t off = 0;
    auto alloc = [&](size_t bytes) -> void* {
        void* r = (void*)((char*)d_ws + off);
        off += (bytes + 255) & ~(size_t)255;
        return r;
    };
    int*   IDX  = (int*)  alloc((size_t)262144 * sizeof(int));      // B*M*k max (L1)
    float* PL   = (float*)alloc((size_t)786432 * sizeof(float));    // rows*3 max
    float* T1   = (float*)alloc((size_t)21233664 * sizeof(float));  // rows*kk max (L2)
    float* T2   = (float*)alloc((size_t)21233664 * sizeof(float));
    float* HCAT = (float*)alloc((size_t)18874368 * sizeof(float));  // rows*C max (L4)
    float* HTR  = (float*)alloc((size_t)18874368 * sizeof(float));
    float* TT   = (float*)alloc((size_t)2097152 * sizeof(float));   // B*M*kk max (L1)
    float* XA   = (float*)alloc((size_t)1572864 * sizeof(float));   // B*M*cout max
    float* XB   = (float*)alloc((size_t)1572864 * sizeof(float));
    float* H1   = (float*)alloc((size_t)4096 * 256 * sizeof(float));
    float* H2   = (float*)alloc((size_t)4096 * 128 * sizeof(float));
    float* LOG  = (float*)alloc((size_t)4096 * 40 * sizeof(float));

    auto gemm = [&](const float* A, int lda, const float* Bm, int ldb, int btrans,
                    const float* bias, const float* g, const float* bb,
                    const float* mm, const float* vv, int act,
                    float* C, int ldc, int M, int N, int K) {
        dim3 gr((N + BNT - 1) / BNT, (M + BMT - 1) / BMT);
        wmma_gemm_kernel<<<gr, 128, 0, stream>>>(A, lda, Bm, ldb, btrans, bias,
                                                 g, bb, mm, vv, act, C, ldc, M, N, K);
    };

    // One XConv layer. Param leaf order within a layer (sorted pytree keys):
    // conv_b, conv_w, mlp_b1, mlp_b2, mlp_bn1[4], mlp_bn2[4], mlp_w1, mlp_w2,
    // stn_b1, stn_b2, stn_b3, stn_bn1[4], stn_bn2[4], stn_w1, stn_w2, stn_w3.
    auto run_layer = [&](int base, int N, int Mq, int k, int dil, int cin, int cout,
                         const float* xin, int Nx, float* xout) {
        const int mid = cout / 4, kk = k * k, Cc = mid + cin;
        const int rows = B * Mq * k;
        const int bm   = B * Mq;
        const float* conv_b = FP(base + 0);
        const float* conv_w = FP(base + 1);
        const float* mlp_b1 = FP(base + 2);
        const float* mlp_b2 = FP(base + 3);
        const float *m1g = FP(base + 4), *m1b = FP(base + 5), *m1m = FP(base + 6), *m1v = FP(base + 7);
        const float *m2g = FP(base + 8), *m2b = FP(base + 9), *m2m = FP(base + 10), *m2v = FP(base + 11);
        const float* mlp_w1 = FP(base + 12);
        const float* mlp_w2 = FP(base + 13);
        const float* s_b1 = FP(base + 14);
        const float* s_b2 = FP(base + 15);
        const float* s_b3 = FP(base + 16);
        const float *s1g = FP(base + 17), *s1b = FP(base + 18), *s1m = FP(base + 19), *s1v = FP(base + 20);
        const float *s2g = FP(base + 21), *s2b = FP(base + 22), *s2m = FP(base + 23), *s2v = FP(base + 24);
        const float* s_w1 = FP(base + 25);
        const float* s_w2 = FP(base + 26);
        const float* s_w3 = FP(base + 27);

        // 1) kNN indices
        knn_kernel<<<(bm + 255) / 256, 256, 0, stream>>>(P, N, Mq, B, k, dil, IDX);
        // 2) gather pl + x features into concat buffer
        gather_kernel<<<(rows + 255) / 256, 256, 0, stream>>>(
            P, IDX, xin, Nx, cin, Mq, B, k, Cc, mid, PL, HCAT);
        // 3) feature MLP layer 1 (K=3), h1 -> T1
        {
            long total = (long)rows * mid;
            dense3_kernel<<<(unsigned)((total + 255) / 256), 256, 0, stream>>>(
                PL, mlp_w1, mlp_b1, m1g, m1b, m1m, m1v, T1, total, mid);
        }
        // 4) feature MLP layer 2: (rows x mid)@(mid x mid) -> hcat[:, 0:mid], ELU+BN
        gemm(T1, mid, mlp_w2, mid, 0, mlp_b2, m2g, m2b, m2m, m2v, 2,
             HCAT, Cc, rows, mid, mid);
        // 5) STN layer 1 (K=3), t1 -> T1
        {
            long total = (long)rows * kk;
            dense3_kernel<<<(unsigned)((total + 255) / 256), 256, 0, stream>>>(
                PL, s_w1, s_b1, s1g, s1b, s1m, s1v, T1, total, kk);
        }
        // 6) STN layer 2: (rows x kk)@(kk x kk) -> T2, ELU+BN
        gemm(T1, kk, s_w2, kk, 0, s_b2, s2g, s2b, s2m, s2v, 2,
             T2, kk, rows, kk, kk);
        // 7) T = einsum('bmkc,kcd->bmd'): (bm x k*kk)@(k*kk x kk) -> TT
        gemm(T2, k * kk, s_w3, kk, 0, s_b3, nullptr, nullptr, nullptr, nullptr, 0,
             TT, kk, bm, kk, k * kk);
        // 8) apply per-point kxk transform
        {
            long total = (long)bm * k * Cc;
            trans_kernel<<<(unsigned)((total + 255) / 256), 256, 0, stream>>>(
                TT, HCAT, HTR, bm, k, Cc);
        }
        // 9) conv: (bm x k*Cc) @ conv_w(cout, k*Cc)^T -> xout
        gemm(HTR, k * Cc, conv_w, k * Cc, 1, conv_b,
             nullptr, nullptr, nullptr, nullptr, 0, xout, cout, bm, cout, k * Cc);
    };

    // Layer param bases: c1=1, c2=29, c3=57, c4=85 (28 leaves each); head=113.
    run_layer(1,  2048, 1024,  8, 1,   0,  48, nullptr,    0, XA);
    run_layer(29, 1024,  384, 12, 2,  48,  96, XA,      1024, XB);
    run_layer(57,  384,  128, 16, 2,  96, 192, XB,       384, XA);
    run_layer(85,  128,  128, 16, 3, 192, 384, XA,       128, XB);

    // Head: leaves b1,b2,bf,bn1[4],bn2[4],w1,w2,wf
    const float *hb1 = FP(113), *hb2 = FP(114), *hbf = FP(115);
    const float *h1g = FP(116), *h1b = FP(117), *h1m = FP(118), *h1v = FP(119);
    const float *h2g = FP(120), *h2b = FP(121), *h2m = FP(122), *h2v = FP(123);
    const float *hw1 = FP(124), *hw2 = FP(125), *hwf = FP(126);
    const int bm = B * 128;
    gemm(XB, 384, hw1, 256, 0, hb1, h1g, h1b, h1m, h1v, 1, H1, 256, bm, 256, 384);
    gemm(H1, 256, hw2, 128, 0, hb2, h2g, h2b, h2m, h2v, 1, H2, 128, bm, 128, 256);
    gemm(H2, 128, hwf, 40, 0, hbf, nullptr, nullptr, nullptr, nullptr, 0,
         LOG, 40, bm, 40, 128);
    mean_kernel<<<(B * 40 + 255) / 256, 256, 0, stream>>>(LOG, (float*)d_out, B, 128, 40);
}
